// AnomalyAttention_34540126994628
// MI455X (gfx1250) — matrix operand order repaired and checked by reading, not compiled, not run on hardware
//
#include <hip/hip_runtime.h>

// ---------------------------------------------------------------------------
// AnomalyAttention forward for MI455X (gfx1250, wave32, WMMA).
// Shapes fixed by the reference: B=8, W=1024, D=512, H=8, HS=64.
//
// Roofline: 553MB of outputs (out+series+prior) at 23.3TB/s dominates; GEMMs
// total only ~35 GFLOP, so they run as bf16 WMMA (f32 accumulate) and scores
// are written straight into the series output region and softmaxed in place.
// ---------------------------------------------------------------------------

#define NB   8
#define NW   1024
#define ND   512
#define NH   8
#define NHS  64
#define NM   (NB * NW)          // 8192 rows
#define LDST 40                 // LDS row stride (ushorts): 32 + 8 pad, keeps 16B align

typedef __attribute__((ext_vector_type(16))) __bf16         bf16x16;
typedef __attribute__((ext_vector_type(8)))  float          floatx8;
typedef __attribute__((ext_vector_type(8)))  unsigned short ushort8;

__device__ __forceinline__ unsigned short f2bf(float f) {
  unsigned u = __float_as_uint(f);
  u += 0x7FFFu + ((u >> 16) & 1u);      // round-to-nearest-even
  return (unsigned short)(u >> 16);
}

union FragU { bf16x16 v; ushort8 u[2]; };

// One wave: 16x16x32 WMMA pair sharing the A fragment (tiles N and N+16).
// A LDS tile: [64 rows M][32 K] bf16, stride LDST.
// B LDS tile: [64 rows N][32 K] bf16 (i.e. stored K-transposed), stride LDST.
__device__ __forceinline__ void wave_tile_mma(const unsigned short* As,
                                              const unsigned short* Bs,
                                              int waveM, int waveN, int lane,
                                              floatx8& c0, floatx8& c1) {
  const int mr = waveM + (lane & 15);
  const int kg = (lane >> 4) << 4;        // A: K 0-15 / 16-31 by lane half
  FragU a;
  a.u[0] = *(const ushort8*)&As[mr * LDST + kg];
  a.u[1] = *(const ushort8*)&As[mr * LDST + kg + 8];
  const int kq = (lane >> 4) << 4;        // B: K 0-15 / 16-31 by lane half
  const int nc = lane & 15;
  FragU b0, b1;
  b0.u[0] = *(const ushort8*)&Bs[(waveN + nc) * LDST + kq];
  b0.u[1] = *(const ushort8*)&Bs[(waveN + nc) * LDST + kq + 8];
  b1.u[0] = *(const ushort8*)&Bs[(waveN + 16 + nc) * LDST + kq];
  b1.u[1] = *(const ushort8*)&Bs[(waveN + 16 + nc) * LDST + kq + 8];
  c0 = __builtin_amdgcn_wmma_f32_16x16x32_bf16(false, a.v, false, b0.v,
                                               (short)0, c0, false, false);
  c1 = __builtin_amdgcn_wmma_f32_16x16x32_bf16(false, a.v, false, b1.v,
                                               (short)0, c1, false, false);
}

// ---------------------------------------------------------------------------
// sigma[b,h,w] = 3^(sigmoid(5*(q[b,w,:]·Wsig[:,h]+bsig[h])) + 1e-5) - 1
// ---------------------------------------------------------------------------
__global__ __launch_bounds__(256) void sigma_kernel(const float* __restrict__ q,
                                                    const float* __restrict__ Wsig,
                                                    const float* __restrict__ bsig,
                                                    float* __restrict__ sigma) {
  const int idx = blockIdx.x * 256 + threadIdx.x;   // 65536 = 8192 rows * 8 heads
  const int h   = idx & 7;
  const int row = idx >> 3;                         // b*W + w
  const float* qr = q + (size_t)row * ND;
  float acc = bsig[h];
  for (int e = 0; e < ND; ++e)
    acc += qr[e] * Wsig[e * NH + h];
  const float sg  = 1.0f / (1.0f + __expf(-5.0f * acc));
  const float val = __expf(1.0986122886681098f * (sg + 1e-5f)) - 1.0f;  // 3^x - 1
  const int b = row >> 10, w = row & 1023;
  sigma[((size_t)(b * NH + h)) * NW + w] = val;
}

// ---------------------------------------------------------------------------
// prior[b,h,i,j] = 1/(sqrt(2pi)*s) * exp(-0.5*((|i-j|)/s)^2)
// ---------------------------------------------------------------------------
__global__ __launch_bounds__(256) void prior_kernel(const float* __restrict__ sigma,
                                                    float* __restrict__ prior) {
  const int bh = blockIdx.x >> 10;
  const int i  = blockIdx.x & 1023;
  const float s    = sigma[(size_t)bh * NW + i];
  const float inv  = 1.0f / s;
  const float coef = 0.3989422804014327f * inv;
  const int j0 = threadIdx.x * 4;
  float4 r;
  {
    float d;
    d = fabsf((float)(i - (j0 + 0))) * inv; r.x = coef * __expf(-0.5f * d * d);
    d = fabsf((float)(i - (j0 + 1))) * inv; r.y = coef * __expf(-0.5f * d * d);
    d = fabsf((float)(i - (j0 + 2))) * inv; r.z = coef * __expf(-0.5f * d * d);
    d = fabsf((float)(i - (j0 + 3))) * inv; r.w = coef * __expf(-0.5f * d * d);
  }
  ((float4*)(prior + (size_t)blockIdx.x * NW))[threadIdx.x] = r;
}

// ---------------------------------------------------------------------------
// Generic [8192x512]x[512x512]+bias GEMM (WMMA bf16, f32 accumulate).
//   ABF16   : A source is bf16 (ushort) else fp32 (converted while staging)
//   OUTMODE : 0 -> fp32 row-major [M,512] (final out projection)
//             1 -> bf16 permuted [B,H,W,HS]  (q/k/v head projections)
// ---------------------------------------------------------------------------
template <bool ABF16, int OUTMODE>
__global__ __launch_bounds__(256) void gemm512_kernel(const void*  __restrict__ Ap,
                                                      const float* __restrict__ Bw,
                                                      const float* __restrict__ bias,
                                                      void* __restrict__ Out) {
  __shared__ __align__(16) unsigned short As[64 * LDST];
  __shared__ __align__(16) unsigned short Bs[64 * LDST];
  const int tid  = threadIdx.x;
  const int lane = tid & 31, wave = tid >> 5;
  const int waveM = (wave & 3) << 4;           // 0..48
  const int waveN = (wave >> 2) << 5;          // 0 or 32
  const int m0 = blockIdx.y << 6;
  const int n0 = blockIdx.x << 6;
  floatx8 c0 = {}, c1 = {};

  const int ra = tid >> 2, ca = (tid & 3) << 3;  // A stage: 64 rows x 32 K
  const int kb = tid >> 3, nb = (tid & 7) << 3;  // B stage: 32 K x 64 N

  for (int k0 = 0; k0 < ND; k0 += 32) {
    if constexpr (ABF16) {
      const unsigned short* A = (const unsigned short*)Ap;
      ushort8 av = *(const ushort8*)(A + (size_t)(m0 + ra) * ND + k0 + ca);
      *(ushort8*)&As[ra * LDST + ca] = av;
    } else {
      const float* A = (const float*)Ap;
      const float4* ap = (const float4*)(A + (size_t)(m0 + ra) * ND + k0 + ca);
      if (k0 + 32 < ND) __builtin_prefetch((const void*)(ap + 8), 0, 1);
      float4 f0 = ap[0], f1 = ap[1];
      ushort8 av;
      av[0] = f2bf(f0.x); av[1] = f2bf(f0.y); av[2] = f2bf(f0.z); av[3] = f2bf(f0.w);
      av[4] = f2bf(f1.x); av[5] = f2bf(f1.y); av[6] = f2bf(f1.z); av[7] = f2bf(f1.w);
      *(ushort8*)&As[ra * LDST + ca] = av;
    }
    {   // B tile [k0..k0+31][n0..n0+63], stored K-transposed in LDS
      const float4* bp = (const float4*)(Bw + (size_t)(k0 + kb) * ND + n0 + nb);
      float4 g0 = bp[0], g1 = bp[1];
      Bs[(nb + 0) * LDST + kb] = f2bf(g0.x);
      Bs[(nb + 1) * LDST + kb] = f2bf(g0.y);
      Bs[(nb + 2) * LDST + kb] = f2bf(g0.z);
      Bs[(nb + 3) * LDST + kb] = f2bf(g0.w);
      Bs[(nb + 4) * LDST + kb] = f2bf(g1.x);
      Bs[(nb + 5) * LDST + kb] = f2bf(g1.y);
      Bs[(nb + 6) * LDST + kb] = f2bf(g1.z);
      Bs[(nb + 7) * LDST + kb] = f2bf(g1.w);
    }
    __syncthreads();
    wave_tile_mma(As, Bs, waveM, waveN, lane, c0, c1);
    __syncthreads();
  }

  const int nc    = lane & 15;
  const int mhalf = (lane >> 4) << 3;
  const float bb0 = bias[n0 + waveN + nc];
  const float bb1 = bias[n0 + waveN + 16 + nc];
  for (int r = 0; r < 8; ++r) {
    const int gm  = m0 + waveM + mhalf + r;
    const int gn0 = n0 + waveN + nc;
    const int gn1 = gn0 + 16;
    const float v0 = c0[r] + bb0;
    const float v1 = c1[r] + bb1;
    if constexpr (OUTMODE == 0) {
      float* O = (float*)Out;
      O[(size_t)gm * ND + gn0] = v0;
      O[(size_t)gm * ND + gn1] = v1;
    } else {
      unsigned short* O = (unsigned short*)Out;   // [B,H,W,HS]
      const int b = gm >> 10, w = gm & 1023;
      const int h0 = gn0 >> 6, hs0 = gn0 & 63;
      const int h1 = gn1 >> 6, hs1 = gn1 & 63;
      O[(((size_t)(b * NH + h0)) * NW + w) * NHS + hs0] = f2bf(v0);
      O[(((size_t)(b * NH + h1)) * NW + w) * NHS + hs1] = f2bf(v1);
    }
  }
}

// ---------------------------------------------------------------------------
// scores[b,h,q,k] = (qh[b,h,q,:]·kh[b,h,k,:]) / 8 -> written to series region
// ---------------------------------------------------------------------------
__global__ __launch_bounds__(256) void scores_kernel(const unsigned short* __restrict__ qh,
                                                     const unsigned short* __restrict__ kh,
                                                     float* __restrict__ series) {
  __shared__ __align__(16) unsigned short As[64 * LDST];
  __shared__ __align__(16) unsigned short Bs[64 * LDST];
  const int tid  = threadIdx.x;
  const int lane = tid & 31, wave = tid >> 5;
  const int waveM = (wave & 3) << 4;
  const int waveN = (wave >> 2) << 5;
  const int bh = blockIdx.z;
  const int m0 = blockIdx.y << 6;
  const int n0 = blockIdx.x << 6;
  const unsigned short* Ab = qh + (size_t)bh * NW * NHS;
  const unsigned short* Bb = kh + (size_t)bh * NW * NHS;
  floatx8 c0 = {}, c1 = {};

  const int ra = tid >> 2, ca = (tid & 3) << 3;
  for (int k0 = 0; k0 < NHS; k0 += 32) {
    ushort8 av = *(const ushort8*)(Ab + (size_t)(m0 + ra) * NHS + k0 + ca);
    *(ushort8*)&As[ra * LDST + ca] = av;
    ushort8 bv = *(const ushort8*)(Bb + (size_t)(n0 + ra) * NHS + k0 + ca);
    *(ushort8*)&Bs[ra * LDST + ca] = bv;     // kh is already [N][K]
    __syncthreads();
    wave_tile_mma(As, Bs, waveM, waveN, lane, c0, c1);
    __syncthreads();
  }

  float* Sb = series + (size_t)bh * NW * NW;
  const int nc    = lane & 15;
  const int mhalf = (lane >> 4) << 3;
  for (int r = 0; r < 8; ++r) {
    const int gm = m0 + waveM + mhalf + r;
    Sb[(size_t)gm * NW + n0 + waveN + nc]      = c0[r] * 0.125f;
    Sb[(size_t)gm * NW + n0 + waveN + 16 + nc] = c1[r] * 0.125f;
  }
}

// ---------------------------------------------------------------------------
// In-place softmax over the last axis of series (65536 rows x 1024)
// ---------------------------------------------------------------------------
__global__ __launch_bounds__(256) void softmax_kernel(float* __restrict__ series) {
  __shared__ float red[256];
  const int tid = threadIdx.x;
  float4* p = (float4*)(series + (size_t)blockIdx.x * NW);
  float4 x = p[tid];
  float m = fmaxf(fmaxf(x.x, x.y), fmaxf(x.z, x.w));
  red[tid] = m;
  __syncthreads();
  for (int off = 128; off > 0; off >>= 1) {
    if (tid < off) red[tid] = fmaxf(red[tid], red[tid + off]);
    __syncthreads();
  }
  const float rowmax = red[0];
  __syncthreads();
  float4 e;
  e.x = __expf(x.x - rowmax); e.y = __expf(x.y - rowmax);
  e.z = __expf(x.z - rowmax); e.w = __expf(x.w - rowmax);
  red[tid] = e.x + e.y + e.z + e.w;
  __syncthreads();
  for (int off = 128; off > 0; off >>= 1) {
    if (tid < off) red[tid] += red[tid + off];
    __syncthreads();
  }
  const float inv = 1.0f / red[0];
  e.x *= inv; e.y *= inv; e.z *= inv; e.w *= inv;
  p[tid] = e;
}

// ---------------------------------------------------------------------------
// attention[b,w,h,:] = series[b,h,w,:] @ vh[b,h,:,:]  -> bf16 ws [B,W,D]
// ---------------------------------------------------------------------------
__global__ __launch_bounds__(256) void attn_kernel(const float* __restrict__ series,
                                                   const unsigned short* __restrict__ vh,
                                                   unsigned short* __restrict__ attn) {
  __shared__ __align__(16) unsigned short As[64 * LDST];
  __shared__ __align__(16) unsigned short Bs[64 * LDST];
  const int tid  = threadIdx.x;
  const int lane = tid & 31, wave = tid >> 5;
  const int waveM = (wave & 3) << 4;
  const int waveN = (wave >> 2) << 5;
  const int bh = blockIdx.y;
  const int m0 = blockIdx.x << 6;
  const float*          Sb = series + ((size_t)bh << 20);
  const unsigned short* Vb = vh + (size_t)bh * NW * NHS;
  floatx8 c0 = {}, c1 = {};

  const int ra = tid >> 2, ca = (tid & 3) << 3;   // A: 64 x 32
  const int kb = tid >> 3, nb = (tid & 7) << 3;   // B: 32 x 64
  for (int k0 = 0; k0 < NW; k0 += 32) {
    {
      const float4* ap = (const float4*)(Sb + (size_t)(m0 + ra) * NW + k0 + ca);
      if (k0 + 32 < NW) __builtin_prefetch((const void*)(ap + 8), 0, 1);
      float4 f0 = ap[0], f1 = ap[1];
      ushort8 av;
      av[0] = f2bf(f0.x); av[1] = f2bf(f0.y); av[2] = f2bf(f0.z); av[3] = f2bf(f0.w);
      av[4] = f2bf(f1.x); av[5] = f2bf(f1.y); av[6] = f2bf(f1.z); av[7] = f2bf(f1.w);
      *(ushort8*)&As[ra * LDST + ca] = av;
    }
    {
      ushort8 bv = *(const ushort8*)(Vb + (size_t)(k0 + kb) * NHS + nb);
      Bs[(nb + 0) * LDST + kb] = bv[0];
      Bs[(nb + 1) * LDST + kb] = bv[1];
      Bs[(nb + 2) * LDST + kb] = bv[2];
      Bs[(nb + 3) * LDST + kb] = bv[3];
      Bs[(nb + 4) * LDST + kb] = bv[4];
      Bs[(nb + 5) * LDST + kb] = bv[5];
      Bs[(nb + 6) * LDST + kb] = bv[6];
      Bs[(nb + 7) * LDST + kb] = bv[7];
    }
    __syncthreads();
    wave_tile_mma(As, Bs, waveM, waveN, lane, c0, c1);
    __syncthreads();
  }

  const int b = bh >> 3, h = bh & 7;
  const int nc    = lane & 15;
  const int mhalf = (lane >> 4) << 3;
  for (int r = 0; r < 8; ++r) {
    const int gm  = m0 + waveM + mhalf + r;         // w index
    const int gn0 = waveN + nc;                     // d within head
    const int gn1 = gn0 + 16;
    unsigned short* Ob = attn + ((size_t)(b * NW + gm)) * ND + h * NHS;
    Ob[gn0] = f2bf(c0[r]);
    Ob[gn1] = f2bf(c1[r]);
  }
}

// ---------------------------------------------------------------------------
extern "C" void kernel_launch(void* const* d_in, const int* in_sizes, int n_in,
                              void* d_out, int out_size, void* d_ws, size_t ws_size,
                              hipStream_t stream) {
  (void)in_sizes; (void)n_in; (void)out_size; (void)ws_size;
  const float* q    = (const float*)d_in[0];
  const float* k    = (const float*)d_in[1];
  const float* v    = (const float*)d_in[2];
  const float* Wq   = (const float*)d_in[3];
  const float* bq   = (const float*)d_in[4];
  const float* Wk   = (const float*)d_in[5];
  const float* bk   = (const float*)d_in[6];
  const float* Wv   = (const float*)d_in[7];
  const float* bv   = (const float*)d_in[8];
  const float* Wsig = (const float*)d_in[9];
  const float* bsig = (const float*)d_in[10];
  const float* Wo   = (const float*)d_in[11];
  const float* bo   = (const float*)d_in[12];

  // workspace carve (~34 MB): qh/kh/vh/attn bf16 + sigma f32
  unsigned short* qh    = (unsigned short*)d_ws;
  unsigned short* kh    = qh + (size_t)NM * ND;
  unsigned short* vh    = kh + (size_t)NM * ND;
  unsigned short* attnb = vh + (size_t)NM * ND;
  float*          sigma = (float*)(attnb + (size_t)NM * ND);

  float* out    = (float*)d_out;                        // [B,W,D]
  float* series = out + (size_t)NM * ND;                // [B,H,W,W]
  float* prior  = series + (size_t)NB * NH * NW * NW;   // [B,H,W,W]

  // sigma head + Gaussian prior (independent of the WMMA chain)
  sigma_kernel<<<dim3(256), dim3(256), 0, stream>>>(q, Wsig, bsig, sigma);
  prior_kernel<<<dim3(NB * NH * NW), dim3(256), 0, stream>>>(sigma, prior);

  // q/k/v head projections -> bf16 [B,H,W,HS]
  gemm512_kernel<false, 1><<<dim3(ND / 64, NM / 64), dim3(256), 0, stream>>>(q, Wq, bq, qh);
  gemm512_kernel<false, 1><<<dim3(ND / 64, NM / 64), dim3(256), 0, stream>>>(k, Wk, bk, kh);
  gemm512_kernel<false, 1><<<dim3(ND / 64, NM / 64), dim3(256), 0, stream>>>(v, Wv, bv, vh);

  // scores -> series region of d_out, softmax in place
  scores_kernel<<<dim3(NW / 64, NW / 64, NB * NH), dim3(256), 0, stream>>>(qh, kh, series);
  softmax_kernel<<<dim3(NB * NH * NW), dim3(256), 0, stream>>>(series);

  // attention = series @ vh -> bf16 [B,W,D]
  attn_kernel<<<dim3(NW / 64, NB * NH), dim3(256), 0, stream>>>(series, vh, attnb);

  // out = attention @ Wo + bo -> d_out
  gemm512_kernel<true, 0><<<dim3(ND / 64, NM / 64), dim3(256), 0, stream>>>(attnb, Wo, bo, out);
}
